// RandomAttention_63075889709234
// MI455X (gfx1250) — matrix operand order repaired and checked by reading, not compiled, not run on hardware
//
#include <hip/hip_runtime.h>
#include <hip/hip_bf16.h>
#include <stdint.h>

typedef __attribute__((ext_vector_type(16))) __bf16 v16bf;
typedef __attribute__((ext_vector_type(8)))  __bf16 v8bf;
typedef __attribute__((ext_vector_type(8)))  float  v8f;
typedef __attribute__((ext_vector_type(4))) unsigned int u32x4;
typedef __attribute__((ext_vector_type(8))) int          i32x8;
typedef __attribute__((ext_vector_type(4))) int          i32x4;

#define S_LEN   2048
#define D_HEAD  64
#define BLOCK_M 64        // queries per block (4 waves x 16 rows)
#define TILE_N  32        // keys per iteration (two 16-key WMMA tiles)
#define NEG_BIG (-1.0e30f)
#define BH_TOT  32        // B*H

union FragBF {
  v16bf v;
  v8bf  h[2];
  __bf16 e[16];
  uint4 q[2];
};

// 8 contiguous floats -> 8 bf16 (packed v_cvt_pk_bf16_f32)
__device__ __forceinline__ v8bf cvt8(const float* p) {
  v8f x = *(const v8f*)p;
  return __builtin_convertvector(x, v8bf);
}

__device__ __forceinline__ unsigned short bfbits(float x) {
  union { __bf16 b; unsigned short s; } u;
  u.b = (__bf16)x;
  return u.s;
}

__device__ __forceinline__ bool mbit(uint2 m, int r) {
  unsigned w = (r < 4) ? m.x : m.y;
  return ((w >> ((r & 3) * 8)) & 0xffu) != 0u;
}

// ---- TDM: DMA one V tile (64 d-rows x 16 u32 bf16-pairs) into LDS ----
// D# per CDNA5 ISA ch.8: group0 = {count/type/addr}, group1 = {dims/strides}
__device__ __forceinline__ void tdm_load_v_tile(const unsigned* gsrc, unsigned lds_byte_addr) {
  unsigned long long ga = (unsigned long long)(uintptr_t)gsrc;
  u32x4 g0;
  g0[0] = 1u;                                                 // count=1, user D#
  g0[1] = lds_byte_addr;                                      // lds_addr[31:0]
  g0[2] = (unsigned)(ga & 0xffffffffu);                       // global_addr lo
  g0[3] = (unsigned)((ga >> 32) & 0x01ffffffu) | 0x80000000u; // addr hi | type=2
  i32x8 g1;
  g1[0] = 0x00020000;                                         // data_size=4B
  g1[1] = (int)((unsigned)(S_LEN / 2) << 16);                 // tensor_dim0 lo16
  g1[2] = (int)((unsigned)D_HEAD << 16);                      // tensor_dim1 lo16
  g1[3] = (int)(16u << 16);                                   // tile_dim0 = 16 u32
  g1[4] = D_HEAD;                                             // tile_dim1 = 64
  g1[5] = S_LEN / 2;                                          // tensor_dim0_stride lo32
  g1[6] = 0;
  g1[7] = 0;
  i32x4 gz = {0, 0, 0, 0};
#if defined(__clang_major__) && (__clang_major__ >= 23)
  i32x8 gz8 = {0, 0, 0, 0, 0, 0, 0, 0};
  __builtin_amdgcn_tensor_load_to_lds(g0, g1, gz, gz, gz8, 0);
#else
  __builtin_amdgcn_tensor_load_to_lds(g0, g1, gz, gz, 0);
#endif
}

// ---- one-time pre-pass: K -> row-major bf16 ----
__global__ __launch_bounds__(256)
void pack_k_bf16(const float* __restrict__ K, unsigned short* __restrict__ Kbf) {
  size_t i = ((size_t)blockIdx.x * 256 + threadIdx.x) * 8;
  v8bf y = cvt8(K + i);
  *(v8bf*)(Kbf + i) = y;
}

// ---- one-time pre-pass: V -> [bh][d][kpair] packed bf16 pairs ----
__global__ __launch_bounds__(256)
void pack_v_pairs(const float* __restrict__ V, unsigned* __restrict__ Vpk) {
  const int kp = blockIdx.x * 256 + threadIdx.x;   // 0 .. S/2-1
  const int d  = blockIdx.y;                       // 0 .. 63
  const int bh = blockIdx.z;                       // 0 .. BH-1
  const float* vb = V + (size_t)bh * S_LEN * D_HEAD;
  float a = vb[(size_t)(2 * kp)     * D_HEAD + d];
  float b = vb[(size_t)(2 * kp + 1) * D_HEAD + d];
  Vpk[((size_t)bh * D_HEAD + d) * (S_LEN / 2) + kp] =
      (unsigned)bfbits(a) | ((unsigned)bfbits(b) << 16);
}

__global__ __launch_bounds__(128)
void flash_masked_attn_kernel(const float* __restrict__ Q,
                              const unsigned short* __restrict__ Kbf,
                              const unsigned* __restrict__ Vpk,
                              const unsigned char* __restrict__ mask,
                              float* __restrict__ O) {
  // double-buffered V tile: [d][kpair] bf16-pairs, filled by TDM
  __shared__ unsigned int vT[2][D_HEAD * (TILE_N / 2)];   // 2 x 4 KB

  const int lane  = threadIdx.x & 31;
  const int wave  = threadIdx.x >> 5;
  const int n     = lane & 15;       // query column (S^T) / matrix col index
  const int khalf = lane >> 4;       // fragment K-half held by this lane

  const int bh    = blockIdx.y;
  const int qbase = blockIdx.x * BLOCK_M;
  const int qwb   = qbase + wave * 16;
  const int qcol  = qwb + n;         // the query this lane owns in S^T

  const float*          Qb = Q   + (size_t)bh * S_LEN * D_HEAD;
  const unsigned short* Kb = Kbf + (size_t)bh * S_LEN * D_HEAD;
  const unsigned*       Vb = Vpk + (size_t)bh * D_HEAD * (S_LEN / 2);
  float*                Ob = O   + (size_t)bh * S_LEN * D_HEAD;

  // ---- Q as B-fragments of (K_tile . Q^T); persists across KV loop ----
  FragBF qb0, qb1;
  {
    const float* qp = Qb + (size_t)qcol * D_HEAD;
    qb0.h[0] = cvt8(qp + 8 * khalf);
    qb0.h[1] = cvt8(qp + 16 + 8 * khalf);
    qb1.h[0] = cvt8(qp + 32 + 8 * khalf);
    qb1.h[1] = cvt8(qp + 48 + 8 * khalf);
  }

  v8f acc0 = {}, acc1 = {}, acc2 = {}, acc3 = {};
  float m_st = NEG_BIG;
  float l_st = 0.f;

  const int ntiles = (qbase + BLOCK_M - 1) / TILE_N + 1;
  const unsigned char* mrow = mask + (size_t)qcol * S_LEN;

  // prologue: DMA tile 0 into buffer 0 (wave 0 owns the TDM queue)
  if (wave == 0) {
    tdm_load_v_tile(Vb, (unsigned)(size_t)(void*)&vT[0][0]);
  }

  for (int j = 0; j < ntiles; ++j) {
    const int kvb = j * TILE_N;
    __syncthreads();  // everyone done reading buf[(j+1)&1] from iter j-1

    // Always issue the "next" DMA (clamped on last iter -> harmless re-load
    // into the dead buffer), then wait for all-but-newest: TDM is in-order
    // per wave, so tile j is guaranteed resident. Non-issuing waves have
    // TENSORcnt==0, so the wait is a NOP for them (no branch needed).
    if (wave == 0) {
      const int nk = (j + 1 < ntiles) ? (kvb + TILE_N) : kvb;
      tdm_load_v_tile(Vb + nk / 2, (unsigned)(size_t)(void*)&vT[(j + 1) & 1][0]);
    }
    __builtin_amdgcn_s_wait_tensorcnt(1);
    __syncthreads();  // tile j visible to all waves

    // ---- S^T = K_tile . Q^T : two 16-key tiles, k-dim 64 in 2 steps ----
    // A-frags come straight from pre-packed bf16 K: two b128 loads per step.
    v8f s0v, s1v;
    {
      FragBF ka;
      const unsigned short* kr0 = Kb + (size_t)(kvb + n) * D_HEAD;
      v8f z = {};
      ka.q[0] = *(const uint4*)(kr0 + 32 + 8 * khalf);
      ka.q[1] = *(const uint4*)(kr0 + 48 + 8 * khalf);
      v8f c = __builtin_amdgcn_wmma_f32_16x16x32_bf16(false, ka.v, false, qb1.v, (short)0, z, false, false);
      ka.q[0] = *(const uint4*)(kr0 +      8 * khalf);
      ka.q[1] = *(const uint4*)(kr0 + 16 + 8 * khalf);
      s0v   = __builtin_amdgcn_wmma_f32_16x16x32_bf16(false, ka.v, false, qb0.v, (short)0, c, false, false);

      const unsigned short* kr1 = Kb + (size_t)(kvb + 16 + n) * D_HEAD;
      ka.q[0] = *(const uint4*)(kr1 + 32 + 8 * khalf);
      ka.q[1] = *(const uint4*)(kr1 + 48 + 8 * khalf);
      c     = __builtin_amdgcn_wmma_f32_16x16x32_bf16(false, ka.v, false, qb1.v, (short)0, z, false, false);
      ka.q[0] = *(const uint4*)(kr1 +      8 * khalf);
      ka.q[1] = *(const uint4*)(kr1 + 16 + 8 * khalf);
      s1v   = __builtin_amdgcn_wmma_f32_16x16x32_bf16(false, ka.v, false, qb0.v, (short)0, c, false, false);
    }

    if (j + 1 < ntiles) {  // prefetch next K tile rows (global_prefetch_b8)
      __builtin_prefetch(Kb + (size_t)(kvb + TILE_N + n) * D_HEAD, 0, 1);
      __builtin_prefetch(Kb + (size_t)(kvb + TILE_N + 16 + n) * D_HEAD, 0, 1);
    }

    // ---- mask + online softmax (reg r = key 16t+8h+r, lane = query) ----
    const uint2 mk0 = *(const uint2*)(mrow + kvb +      8 * khalf);
    const uint2 mk1 = *(const uint2*)(mrow + kvb + 16 + 8 * khalf);

    const float scale = 0.125f;  // 1/sqrt(64)
    float sa[8], sb[8];
    float tmax = NEG_BIG;
    #pragma unroll
    for (int r = 0; r < 8; ++r) {
      sa[r] = mbit(mk0, r) ? s0v[r] * scale : NEG_BIG;
      sb[r] = mbit(mk1, r) ? s1v[r] * scale : NEG_BIG;
      tmax = fmaxf(tmax, fmaxf(sa[r], sb[r]));
    }
    tmax = fmaxf(tmax, __shfl_xor(tmax, 16, 32));   // other 16 keys in lane^16

    const float mnew  = fmaxf(m_st, tmax);
    const float alpha = __expf(m_st - mnew);
    FragBF pa;
    float ps = 0.f;
    #pragma unroll
    for (int r = 0; r < 8; ++r) {
      float p0 = __expf(sa[r] - mnew);
      float p1 = __expf(sb[r] - mnew);
      ps += p0 + p1;
      pa.e[r]     = (__bf16)p0;   // A-layout elems 0-7  <-> keys 8h+r
      pa.e[8 + r] = (__bf16)p1;   // A-layout elems 8-15 <-> keys 16+8h+r
    }
    ps += __shfl_xor(ps, 16, 32);
    l_st = l_st * alpha + ps;
    m_st = mnew;

    // redistribute alpha to acc rows (acc reg r holds query r + 8*khalf)
    #pragma unroll
    for (int r = 0; r < 8; ++r) {
      const float ar = __shfl(alpha, r + 8 * khalf, 32);
      acc0[r] *= ar; acc1[r] *= ar; acc2[r] *= ar; acc3[r] *= ar;
    }

    // ---- O += P . V  (P already in A-layout; V B-frags from TDM'd LDS) ----
    const unsigned int* vbuf = &vT[j & 1][0];
    #pragma unroll
    for (int dt = 0; dt < 4; ++dt) {
      FragBF vbf;
      const uint4* vp = (const uint4*)&vbuf[(dt * 16 + n) * (TILE_N / 2) + khalf * 4];
      vbf.q[0] = vp[0];
      vbf.q[1] = vp[2];
      v8f* accp = (dt == 0) ? &acc0 : (dt == 1) ? &acc1 : (dt == 2) ? &acc2 : &acc3;
      *accp = __builtin_amdgcn_wmma_f32_16x16x32_bf16(false, pa.v, false, vbf.v, (short)0, *accp, false, false);
    }
  }

  // ---- epilogue: fetch this row's denom, normalize, store fp32 ----
  #pragma unroll
  for (int r = 0; r < 8; ++r) {
    const float lr  = __shfl(l_st, r + 8 * khalf, 32);
    const float inv = 1.0f / lr;
    const size_t ro = (size_t)(qwb + r + 8 * khalf) * D_HEAD;
    Ob[ro +  0 + n] = acc0[r] * inv;
    Ob[ro + 16 + n] = acc1[r] * inv;
    Ob[ro + 32 + n] = acc2[r] * inv;
    Ob[ro + 48 + n] = acc3[r] * inv;
  }
}

extern "C" void kernel_launch(void* const* d_in, const int* in_sizes, int n_in,
                              void* d_out, int out_size, void* d_ws, size_t ws_size,
                              hipStream_t stream) {
  (void)in_sizes; (void)n_in; (void)out_size; (void)ws_size;
  const float*         q  = (const float*)d_in[0];
  const float*         k  = (const float*)d_in[1];
  const float*         v  = (const float*)d_in[2];
  const unsigned char* mk = (const unsigned char*)d_in[3];  // jax bool -> 1 byte
  float*               o  = (float*)d_out;

  const size_t kelems = (size_t)BH_TOT * S_LEN * D_HEAD;    // 4.19M
  unsigned short* kbf = (unsigned short*)d_ws;              // 8.39 MB
  unsigned*       vpk = (unsigned*)((char*)d_ws + kelems * sizeof(unsigned short));

  // one-time precision/layout pre-passes
  pack_k_bf16 <<<dim3((unsigned)(kelems / (256 * 8))), dim3(256), 0, stream>>>(k, kbf);
  pack_v_pairs<<<dim3(S_LEN / 2 / 256, D_HEAD, BH_TOT), dim3(256), 0, stream>>>(v, vpk);

  dim3 grid(S_LEN / BLOCK_M, BH_TOT);
  dim3 block(128);
  flash_masked_attn_kernel<<<grid, block, 0, stream>>>(q, kbf, vpk, mk, o);
}
